// ObjectRepresentationSNN_16776142258624
// MI455X (gfx1250) — matrix operand order repaired and checked by reading, not compiled, not run on hardware
//
#include <hip/hip_runtime.h>

// ---------------- problem constants (match reference) ----------------
constexpr int   kB        = 64;
constexpr int   kN        = 12288;     // H*W*C
constexpr int   kD        = 8;
constexpr int   kSteps    = 32;
constexpr int   kClsStart = 16;
constexpr int   kClasses  = 10;
constexpr float kFb       = 0.5f;      // FB_AFFINITY
constexpr float kDt       = 0.1f;
constexpr float kDecay    = 0.9f;
constexpr float kThresh   = 0.5f;
constexpr float kRec      = 0.1f;
constexpr int   kBlk      = 256;       // threads per block (8 wave32)
constexpr int   kNBlk     = kN / kBlk; // 48 blocks per batch image

typedef __attribute__((ext_vector_type(16))) _Float16 v16h;
typedef __attribute__((ext_vector_type(8)))  float    v8f;

__device__ __forceinline__ float tanh_fast(float z) {
  float a = fabsf(z);
  float e = __expf(-2.0f * a);
  float t = (1.0f - e) / (1.0f + e);
  return copysignf(t, z);
}

__device__ __forceinline__ float wave_reduce(float v) {
#pragma unroll
  for (int m = 16; m >= 1; m >>= 1) v += __shfl_xor(v, m, 32);
  return v;
}

// ---------------------------------------------------------------------
// init: gamma0 from image, theta0 = normalize(gamma0+1e-3), fill both
// delay-ring slots, zero v/s/rate, emit per-block partial sums of theta0
// (mean-field m for step 0); ms partials are zero (s0 == 0).
// grid = (kNBlk, kB), block = 256
// ---------------------------------------------------------------------
__global__ void init_kernel(const float* __restrict__ x,
                            const float* __restrict__ w_init,
                            const float* __restrict__ b_init,
                            const float* __restrict__ gain,
                            float* __restrict__ theta,
                            float* __restrict__ ring0,
                            float* __restrict__ ring1,
                            float* __restrict__ v,
                            float* __restrict__ s,
                            float* __restrict__ rate,
                            float* __restrict__ pm,
                            float* __restrict__ pms) {
  __shared__ float sw[kD], sb[kD];
  __shared__ float red[8][kD];   // [wave][d]
  const int tid = threadIdx.x;
  if (tid < kD) { sw[tid] = w_init[tid]; sb[tid] = b_init[tid]; }
  __syncthreads();

  const int b    = blockIdx.y;
  const int node = blockIdx.x * kBlk + tid;
  const size_t nidx = (size_t)b * kN + node;
  const size_t base = nidx * kD;

  const float xv = x[nidx];
  const float* gg = gain + (size_t)node * kD;
  float ggr[kD];
  {
    const float4* p = (const float4*)gg;
    float4 a = p[0], c = p[1];
    ggr[0]=a.x; ggr[1]=a.y; ggr[2]=a.z; ggr[3]=a.w;
    ggr[4]=c.x; ggr[5]=c.y; ggr[6]=c.z; ggr[7]=c.w;
  }

  float th[kD]; float nrm = 0.f;
#pragma unroll
  for (int d = 0; d < kD; ++d) {
    float g0 = tanh_fast(fabsf(xv * sw[d] + sb[d]) * ggr[d]);
    th[d] = g0 + 1e-3f;
    nrm += th[d] * th[d];
  }
  const float inv = 1.0f / (sqrtf(nrm) + 1e-6f);
#pragma unroll
  for (int d = 0; d < kD; ++d) th[d] *= inv;

  const float4 lo = make_float4(th[0], th[1], th[2], th[3]);
  const float4 hi = make_float4(th[4], th[5], th[6], th[7]);
  ((float4*)(theta + base))[0] = lo; ((float4*)(theta + base))[1] = hi;
  ((float4*)(ring0 + base))[0] = lo; ((float4*)(ring0 + base))[1] = hi;
  ((float4*)(ring1 + base))[0] = lo; ((float4*)(ring1 + base))[1] = hi;
  v[nidx] = 0.f; s[nidx] = 0.f; rate[nidx] = 0.f;

  // deterministic block partial of sum(theta0) over the 256 nodes here
  const int lane = tid & 31, wv = tid >> 5;
#pragma unroll
  for (int d = 0; d < kD; ++d) {
    float r = wave_reduce(th[d]);
    if (lane == 0) red[wv][d] = r;
  }
  __syncthreads();
  if (tid < kD) {
    float acc = 0.f;
#pragma unroll
    for (int w = 0; w < 8; ++w) acc += red[w][tid];
    const size_t pi = ((size_t)b * kNBlk + blockIdx.x) * kD + tid;
    pm[pi]  = acc;
    pms[pi] = 0.f;   // s0 == 0
  }
}

// ---------------------------------------------------------------------
// deterministic partial->mean-field slot reduction (fixed order)
// one thread per (b,d): 512 threads total
// ---------------------------------------------------------------------
__global__ void reduce_partials_kernel(const float* __restrict__ pm,
                                       const float* __restrict__ pms,
                                       float* __restrict__ mo,
                                       float* __restrict__ mso) {
  const int i = blockIdx.x * blockDim.x + threadIdx.x;
  if (i >= kB * kD) return;
  const int b = i >> 3, d = i & 7;
  float a = 0.f, c = 0.f;
  for (int blk = 0; blk < kNBlk; ++blk) {
    const size_t pi = ((size_t)b * kNBlk + blk) * kD + d;
    a += pm[pi];
    c += pms[pi];
  }
  mo[i] = a; mso[i] = c;
}

// ---------------------------------------------------------------------
// one Kuramoto/SNN step. gamma is recomputed from theta (never stored),
// delay handled via ring slot (t & 1): at step t the slot holds theta_{t-1}
// (theta0 for t<2); we overwrite it with theta_{t+1} for use at step t+2.
// grid = (kNBlk, kB), block = 256
// ---------------------------------------------------------------------
__global__ void step_kernel(int t,
                            const float* __restrict__ x,
                            const float* __restrict__ w_init,
                            const float* __restrict__ b_init,
                            const float* __restrict__ w_read,
                            const float* __restrict__ b_read,
                            const float* __restrict__ gain,
                            const float* __restrict__ w_in,
                            float* __restrict__ theta,
                            float* __restrict__ ring,
                            float* __restrict__ v,
                            float* __restrict__ s,
                            float* __restrict__ rate,
                            const float* __restrict__ m_in,   // raw sums, slot t
                            const float* __restrict__ ms_in,
                            float* __restrict__ pm,           // partials for slot t+1
                            float* __restrict__ pms) {
  __shared__ float s_wr[kD * kD];
  __shared__ float s_br[kD], s_wi[kD], s_bi[kD], s_win[kD];
  __shared__ float s_m[kD], s_ms[kD];
  __shared__ float red_m[8][kD], red_ms[8][kD];

  const int tid = threadIdx.x;
  const int b   = blockIdx.y;
  if (tid < kD * kD) s_wr[tid] = w_read[tid];
  if (tid < kD) {
    s_br[tid]  = b_read[tid];
    s_wi[tid]  = w_init[tid];
    s_bi[tid]  = b_init[tid];
    s_win[tid] = w_in[tid];
    const float invN = 1.0f / (float)kN;
    s_m[tid]  = m_in [b * kD + tid] * invN;   // mean over nodes
    s_ms[tid] = ms_in[b * kD + tid] * invN;
  }
  __syncthreads();

  const int node = blockIdx.x * kBlk + tid;
  const size_t nidx = (size_t)b * kN + node;
  const size_t base = nidx * kD;

  float th[kD];
  {
    const float4* p = (const float4*)(theta + base);
    float4 a = p[0], c = p[1];
    th[0]=a.x; th[1]=a.y; th[2]=a.z; th[3]=a.w;
    th[4]=c.x; th[5]=c.y; th[6]=c.z; th[7]=c.w;
  }
  const float sv = s[nidx];
  const float vv = v[nidx];
  float ggr[kD];
  {
    const float4* p = (const float4*)(gain + (size_t)node * kD);
    float4 a = p[0], c = p[1];
    ggr[0]=a.x; ggr[1]=a.y; ggr[2]=a.z; ggr[3]=a.w;
    ggr[4]=c.x; ggr[5]=c.y; ggr[6]=c.z; ggr[7]=c.w;
  }

  // gamma at start of step: step 0 -> from image (abs BEFORE gain, per ref);
  // else recompute gamma_t = tanh(|(theta_t @ W_read + b_read) * gain|).
  float g[kD];
  if (t == 0) {
    const float xv = x[nidx];
#pragma unroll
    for (int d = 0; d < kD; ++d)
      g[d] = tanh_fast(fabsf(xv * s_wi[d] + s_bi[d]) * ggr[d]);
  } else {
#pragma unroll
    for (int d = 0; d < kD; ++d) {
      float acc = s_br[d];
#pragma unroll
      for (int k = 0; k < kD; ++k) acc += th[k] * s_wr[k * kD + d];
      g[d] = tanh_fast(fabsf(acc * ggr[d]));
    }
  }

  // Kuramoto tangent update with rank-1 spike affinity
  float dr[kD]; float dot = 0.f;
#pragma unroll
  for (int d = 0; d < kD; ++d) {
    dr[d] = (s_m[d] + kFb * sv * s_ms[d]) + g[d];   // COUPLING=ATTRACTION=1
    dot  += dr[d] * th[d];
  }
  float tn[kD]; float nrm = 0.f;
#pragma unroll
  for (int d = 0; d < kD; ++d) {
    tn[d] = th[d] + kDt * (dr[d] - dot * th[d]);
    nrm  += tn[d] * tn[d];
  }
  const float inv = 1.0f / (sqrtf(nrm) + 1e-6f);
#pragma unroll
  for (int d = 0; d < kD; ++d) tn[d] *= inv;

  // delayed gate from ring slot (theta from DELAY steps back)
  float msum;
  {
    const float4* p = (const float4*)(ring + base);
    float4 a = p[0], c = p[1];
    msum = a.x + a.y + a.z + a.w + c.x + c.y + c.z + c.w;
  }
  const float gate = 0.5f * (1.0f + __sinf(msum * (1.0f / kD)));

  // commit theta_new to state and into the ring slot (read-before-write above)
  const float4 lo = make_float4(tn[0], tn[1], tn[2], tn[3]);
  const float4 hi = make_float4(tn[4], tn[5], tn[6], tn[7]);
  ((float4*)(theta + base))[0] = lo; ((float4*)(theta + base))[1] = hi;
  ((float4*)(ring  + base))[0] = lo; ((float4*)(ring  + base))[1] = hi;

  // gamma_new readout -> input current
  float cur = 0.f;
#pragma unroll
  for (int d = 0; d < kD; ++d) {
    float acc = s_br[d];
#pragma unroll
    for (int k = 0; k < kD; ++k) acc += tn[k] * s_wr[k * kD + d];
    cur += tanh_fast(fabsf(acc * ggr[d])) * s_win[d];
  }
  cur *= gate;

  // LIF membrane / spike (straight-through forward value is the hard spike)
  const float vn = kDecay * vv + cur + kRec * sv;
  const float sn = (vn > kThresh) ? 1.0f : 0.0f;
  v[nidx] = vn * (1.0f - sn);
  s[nidx] = sn;
  if (t >= kClsStart) rate[nidx] += sn;

  // deterministic block partials for the next step's mean field
  const int lane = tid & 31, wv = tid >> 5;
#pragma unroll
  for (int d = 0; d < kD; ++d) {
    float rm  = wave_reduce(tn[d]);
    float rms = wave_reduce(sn * tn[d]);
    if (lane == 0) { red_m[wv][d] = rm; red_ms[wv][d] = rms; }
  }
  __syncthreads();
  if (tid < kD) {
    float am = 0.f, as = 0.f;
#pragma unroll
    for (int w = 0; w < 8; ++w) { am += red_m[w][tid]; as += red_ms[w][tid]; }
    const size_t pi = ((size_t)b * kNBlk + blockIdx.x) * kD + tid;
    pm[pi] = am; pms[pi] = as;
  }
}

// ---------------------------------------------------------------------
// rate (spike counts over 16 steps) -> f16 mean rate; k/16 is exact in f16
// ---------------------------------------------------------------------
__global__ void rate_to_f16_kernel(const float* __restrict__ rate,
                                   _Float16* __restrict__ rate16) {
  const int i = blockIdx.x * blockDim.x + threadIdx.x;
  rate16[i] = (_Float16)(rate[i] * (1.0f / 16.0f));
}

// ---------------------------------------------------------------------
// prepack w_cls [12288,10] -> f16 B-fragments in the exact
// V_WMMA_F32_16X16X32_F16 lane layout (classes padded to 16):
// lane L holds col c=L&15, K = k0 + ((L>>4)&1)*8 + 2j + (j>=4 ? 8 : 0) + {0,1} in dword j.
// ---------------------------------------------------------------------
__global__ void prepack_w_kernel(const float* __restrict__ w_cls,
                                 unsigned int* __restrict__ wfrag) {
  const int idx = blockIdx.x * blockDim.x + threadIdx.x;  // kN*8 dwords total
  if (idx >= kN * 8) return;
  const int j    = idx & 7;
  const int lane = (idx >> 3) & 31;
  const int kt   = idx >> 8;
  const int c    = lane & 15;
  const int k    = kt * 32 + ((lane >> 4) & 1) * 8 + 2 * j + ((j >> 2) << 3);
  _Float16 h0 = (c < kClasses) ? (_Float16)w_cls[(size_t)k * kClasses + c] : (_Float16)0.0f;
  _Float16 h1 = (c < kClasses) ? (_Float16)w_cls[(size_t)(k + 1) * kClasses + c] : (_Float16)0.0f;
  union { _Float16 h; unsigned short u; } p0{h0}, p1{h1};
  wfrag[idx] = (unsigned)p0.u | ((unsigned)p1.u << 16);
}

// ---------------------------------------------------------------------
// classifier GEMM: logits[64,10] = rate[64,12288] @ w_cls + b_cls
// 4 blocks x 1 wave32, one 16x16 output tile each, K looped 384x over
// v_wmma_f32_16x16x32_f16. EXEC is all-ones inside the wave (no divergence
// before the WMMA loop).
// A-frag per lane: row = Mtile*16 + (lane&15); two contiguous 16B chunks at
// K offsets {0,16} (lanes 0-15) / {8,24} (lanes 16-31) per ISA layout.
// ---------------------------------------------------------------------
__global__ void cls_gemm_kernel(const _Float16* __restrict__ rate16,
                                const unsigned int* __restrict__ wfrag,
                                const float* __restrict__ b_cls,
                                float* __restrict__ out) {
  const int mtile = blockIdx.x;       // 0..3 (rows of B=64 in 16-chunks)
  const int lane  = threadIdx.x;      // 0..31
  const int arow  = mtile * 16 + (lane & 15);
  const int asel  = (lane & 16) ? 8 : 0;
  const _Float16* aptr = rate16 + (size_t)arow * kN;

  union AU { v16h v; float4 f[2]; };
  union BU { v16h v; uint4  u[2]; };
  v8f acc = {};

  for (int kt = 0; kt < kN / 32; ++kt) {
    AU a;
    const float4* pa = (const float4*)(aptr + kt * 32 + asel);
    a.f[0] = pa[0];            // K = base .. base+7
    a.f[1] = pa[2];            // K = base+16 .. base+23
    BU bm;
    const uint4* pb = (const uint4*)(wfrag + ((size_t)(kt * 32 + lane)) * 8);
    bm.u[0] = pb[0]; bm.u[1] = pb[1];
    acc = __builtin_amdgcn_wmma_f32_16x16x32_f16(
        /*neg_a=*/false, a.v, /*neg_b=*/false, bm.v,
        /*c_mod=*/(short)0, acc, /*reuse_a=*/false, /*reuse_b=*/false);
  }

  const int col = lane & 15;
#pragma unroll
  for (int i = 0; i < 8; ++i) {
    const int row = mtile * 16 + i + ((lane & 16) ? 8 : 0);
    if (col < kClasses) out[row * kClasses + col] = acc[i] + b_cls[col];
  }
}

// ---------------------------------------------------------------------
extern "C" void kernel_launch(void* const* d_in, const int* in_sizes, int n_in,
                              void* d_out, int out_size, void* d_ws, size_t ws_size,
                              hipStream_t stream) {
  (void)in_sizes; (void)n_in; (void)out_size; (void)ws_size;
  const float* x      = (const float*)d_in[0];
  const float* w_init = (const float*)d_in[1];
  const float* b_init = (const float*)d_in[2];
  const float* w_read = (const float*)d_in[3];
  const float* b_read = (const float*)d_in[4];
  const float* gain   = (const float*)d_in[5];
  const float* w_in   = (const float*)d_in[6];
  const float* w_cls  = (const float*)d_in[7];
  const float* b_cls  = (const float*)d_in[8];
  float* out = (float*)d_out;

  // workspace carve-up (~87 MB total; state is L2-resident on MI455X)
  char* ws = (char*)d_ws;
  const size_t SZ = (size_t)kB * kN * kD * sizeof(float);   // 25.2 MB
  const size_t SV = (size_t)kB * kN * sizeof(float);        //  3.1 MB
  const size_t SM = (size_t)(kSteps + 1) * kB * kD * sizeof(float);
  const size_t SP = (size_t)kB * kNBlk * kD * sizeof(float);
  float* theta = (float*)ws; ws += SZ;
  float* ring0 = (float*)ws; ws += SZ;
  float* ring1 = (float*)ws; ws += SZ;
  float* v     = (float*)ws; ws += SV;
  float* s     = (float*)ws; ws += SV;
  float* rate  = (float*)ws; ws += SV;
  float* macc  = (float*)ws; ws += SM;   // raw sums of theta per step slot
  float* msacc = (float*)ws; ws += SM;   // raw sums of s*theta
  float* pm    = (float*)ws; ws += SP;   // per-block partials
  float* pms   = (float*)ws; ws += SP;
  _Float16* rate16 = (_Float16*)ws; ws += (size_t)kB * kN * sizeof(_Float16);
  unsigned* wfrag  = (unsigned*)ws;  ws += (size_t)kN * 8 * sizeof(unsigned);

  const dim3 grid(kNBlk, kB);

  init_kernel<<<grid, kBlk, 0, stream>>>(x, w_init, b_init, gain,
                                         theta, ring0, ring1, v, s, rate, pm, pms);
  reduce_partials_kernel<<<2, 256, 0, stream>>>(pm, pms, macc, msacc); // slot 0

  for (int t = 0; t < kSteps; ++t) {
    float* ring = (t & 1) ? ring1 : ring0;
    step_kernel<<<grid, kBlk, 0, stream>>>(t, x, w_init, b_init, w_read, b_read,
                                           gain, w_in, theta, ring, v, s, rate,
                                           macc  + (size_t)t * kB * kD,
                                           msacc + (size_t)t * kB * kD,
                                           pm, pms);
    reduce_partials_kernel<<<2, 256, 0, stream>>>(pm, pms,
                                                  macc  + (size_t)(t + 1) * kB * kD,
                                                  msacc + (size_t)(t + 1) * kB * kD);
  }

  rate_to_f16_kernel<<<(kB * kN) / 256, 256, 0, stream>>>(rate, rate16);
  prepack_w_kernel<<<(kN * 8 + 255) / 256, 256, 0, stream>>>(w_cls, wfrag);
  cls_gemm_kernel<<<4, 32, 0, stream>>>(rate16, wfrag, b_cls, out);
}
